// DiscriminationLoss_87806311400035
// MI455X (gfx1250) — matrix operand order repaired and checked by reading, not compiled, not run on hardware
//
#include <hip/hip_runtime.h>
#include <math.h>

typedef __attribute__((ext_vector_type(16))) _Float16 v16h;
typedef __attribute__((ext_vector_type(8)))  float    v8f;

#define PPX             (640 * 640)     // pixels per image
#define NBATCH          16
#define NBINS           16              // labels 1..16 (label 0 unused by loss)
#define BLOCKS_X        100
#define WAVES_PB        8               // 256 threads = 8 wave32
#define WAVES_PER_BATCH (BLOCKS_X * WAVES_PB)   // 800
#define MT_PER_BATCH    (PPX / 128)             // 3200 macro-tiles (128 px each)
#define MT_PER_WAVE     (MT_PER_BATCH / WAVES_PER_BATCH) // 4, exact

// ---------------------------------------------------------------------------
// Main streaming kernel: per-pixel squared norm -> segment sum + cardinality
// via one-hot WMMA (f16 A/B, f32 accumulate), hierarchical reduction.
// 4 independent accumulators (one per 32-px sub-tile) avoid WMMA->WMMA RAW
// NOP bubbles and allow back-to-back matrix issue with co-executing VALU.
// ---------------------------------------------------------------------------
__global__ __launch_bounds__(256, 1)
void disc_loss_main(const float* __restrict__ pred,
                    const int*   __restrict__ labels,
                    float* __restrict__ g_ss,
                    float* __restrict__ g_card,
                    int*   __restrict__ g_nk)
{
    __shared__ __align__(32) unsigned short s_lab[WAVES_PB][128];
    __shared__ __align__(32) _Float16       s_hi [WAVES_PB][128];
    __shared__ __align__(32) _Float16       s_lo [WAVES_PB][128];
    __shared__ float s_accss[NBINS];
    __shared__ float s_acccd[NBINS];
    __shared__ int   s_nk;

    const int  b      = blockIdx.y;
    const int  tid    = threadIdx.x;
    const int  lane   = tid & 31;
    const int  wv     = tid >> 5;
    const int  gw     = blockIdx.x * WAVES_PB + wv;   // 0..799 within batch
    const int  n      = lane & 15;                     // D/B column this lane owns
    const bool hiHalf = lane >= 16;
    const unsigned short rowv = (unsigned short)(n + 1); // bin row n <-> label n+1

    if (tid < NBINS) { s_accss[tid] = 0.0f; s_acccd[tid] = 0.0f; }
    if (tid == 0)    { s_nk = 0; }
    __syncthreads();

    const size_t labBase  = (size_t)b * PPX;
    const size_t predBase = (size_t)b * 4 * PPX;

    // Loop-invariant B columns: col 1 = ones (cardinality), cols >=3 = zero.
    // Lanes owning cols 0 / 2 overwrite from LDS (sq_hi / sq_lo) per sub-tile.
    v16h bconst;
#pragma unroll
    for (int e = 0; e < 16; ++e)
        bconst[e] = (n == 1) ? (_Float16)1.0f : (_Float16)0.0f;

    v8f acc0, acc1, acc2, acc3;
#pragma unroll
    for (int r = 0; r < 8; ++r) { acc0[r] = 0.0f; acc1[r] = 0.0f; acc2[r] = 0.0f; acc3[r] = 0.0f; }

    int maxLab = 0;

    for (int it = 0; it < MT_PER_WAVE; ++it) {
        const int mt = gw + it * WAVES_PER_BATCH;
        const int p0 = mt * 128 + lane * 4;

        if (it + 1 < MT_PER_WAVE) {
            const int pn = (gw + (it + 1) * WAVES_PER_BATCH) * 128 + lane * 4;
            __builtin_prefetch(labels + labBase + pn, 0, 1);
            __builtin_prefetch(pred + predBase + 0 * PPX + pn, 0, 1);
            __builtin_prefetch(pred + predBase + 1 * PPX + pn, 0, 1);
            __builtin_prefetch(pred + predBase + 2 * PPX + pn, 0, 1);
            __builtin_prefetch(pred + predBase + 3 * PPX + pn, 0, 1);
        }

        // b128 coalesced loads: 4 pixels per lane, 4 channels + labels
        const int4   lb4 = *(const int4  *)(labels + labBase + p0);
        const float4 x0  = *(const float4*)(pred + predBase + 0 * PPX + p0);
        const float4 x1  = *(const float4*)(pred + predBase + 1 * PPX + p0);
        const float4 x2  = *(const float4*)(pred + predBase + 2 * PPX + p0);
        const float4 x3  = *(const float4*)(pred + predBase + 3 * PPX + p0);

        float sq[4];
        sq[0] = x0.x * x0.x + x1.x * x1.x + x2.x * x2.x + x3.x * x3.x;
        sq[1] = x0.y * x0.y + x1.y * x1.y + x2.y * x2.y + x3.y * x3.y;
        sq[2] = x0.z * x0.z + x1.z * x1.z + x2.z * x2.z + x3.z * x3.z;
        sq[3] = x0.w * x0.w + x1.w * x1.w + x2.w * x2.w + x3.w * x3.w;

        int lb[4];
        lb[0] = lb4.x; lb[1] = lb4.y; lb[2] = lb4.z; lb[3] = lb4.w;

        // Stage labels + hi/lo split of sq (double-f16 ~= f32 precision)
#pragma unroll
        for (int j = 0; j < 4; ++j) {
            const _Float16 h = (_Float16)sq[j];
            const _Float16 l = (_Float16)(sq[j] - (float)h);
            s_lab[wv][lane * 4 + j] = (unsigned short)lb[j];
            s_hi [wv][lane * 4 + j] = h;
            s_lo [wv][lane * 4 + j] = l;
            maxLab = (lb[j] > maxLab) ? lb[j] : maxLab;
        }

        // Same-wave LDS RAW: force completion + stop compiler reordering
        asm volatile("s_wait_dscnt 0x0" ::: "memory");

        // 4 sub-tiles of 32 pixels -> 4 independent WMMAs (no RAW chain)
#pragma unroll
        for (int s = 0; s < 4; ++s) {
            const int base = s * 32;
            // A one-hot: lane holds row M=n; K contiguous per lane half (k = 16h + e), same as B
            const int o0 = base + (hiHalf ? 16 : 0);       // K 0-7 / 16-23
            const int o1 = base + 8 + (hiHalf ? 16 : 0);   // K 8-15 / 24-31
            v16h a;
#pragma unroll
            for (int e = 0; e < 8; ++e) {
                a[e]     = (s_lab[wv][o0 + e] == rowv) ? (_Float16)1.0f : (_Float16)0.0f;
                a[8 + e] = (s_lab[wv][o1 + e] == rowv) ? (_Float16)1.0f : (_Float16)0.0f;
            }
            // B: col 0 = sq_hi, col 1 = ones, col 2 = sq_lo, rest = 0
            const int koff = base + (hiHalf ? 16 : 0);     // K 0-15 / 16-31
            v16h bm = bconst;
            if (n == 0)      bm = *(const v16h*)&s_hi[wv][koff];
            else if (n == 2) bm = *(const v16h*)&s_lo[wv][koff];

            if (s == 0)
                acc0 = __builtin_amdgcn_wmma_f32_16x16x32_f16(false, a, false, bm, (short)0, acc0, false, false);
            else if (s == 1)
                acc1 = __builtin_amdgcn_wmma_f32_16x16x32_f16(false, a, false, bm, (short)0, acc1, false, false);
            else if (s == 2)
                acc2 = __builtin_amdgcn_wmma_f32_16x16x32_f16(false, a, false, bm, (short)0, acc2, false, false);
            else
                acc3 = __builtin_amdgcn_wmma_f32_16x16x32_f16(false, a, false, bm, (short)0, acc3, false, false);
        }
    }

    // Merge the 4 accumulators
    v8f acc;
#pragma unroll
    for (int r = 0; r < 8; ++r) acc[r] = (acc0[r] + acc1[r]) + (acc2[r] + acc3[r]);

    // D layout: lane = column n, VGPR r = row r (+8 for lanes 16-31)
    const int rowOff = hiHalf ? 8 : 0;
    if (n == 0 || n == 2) {          // hi + lo partial squared-sums
#pragma unroll
        for (int r = 0; r < 8; ++r) atomicAdd(&s_accss[rowOff + r], acc[r]);
    } else if (n == 1) {             // cardinalities
#pragma unroll
        for (int r = 0; r < 8; ++r) atomicAdd(&s_acccd[rowOff + r], acc[r]);
    }
    atomicMax(&s_nk, maxLab);
    __syncthreads();

    if (tid < NBINS) {
        atomicAdd(&g_ss  [b * NBINS + tid], s_accss[tid]);
        atomicAdd(&g_card[b * NBINS + tid], s_acccd[tid]);
    }
    if (tid == 0) atomicMax(&g_nk[b], s_nk);
}

// ---------------------------------------------------------------------------
// Finalize: m = ss/card^2; pairwise log1p((3 - sqrt(m_i+m_j))^2); scale; sum.
// ---------------------------------------------------------------------------
__global__ __launch_bounds__(256, 1)
void disc_loss_finalize(const float* __restrict__ g_ss,
                        const float* __restrict__ g_card,
                        const int*   __restrict__ g_nk,
                        float* __restrict__ out)
{
    __shared__ float s_m   [NBATCH][NBINS];
    __shared__ int   s_ok  [NBATCH][NBINS];
    __shared__ float s_part[NBATCH][NBINS];
    __shared__ float s_loss[NBATCH];

    const int tid = threadIdx.x;
    const int b   = tid >> 4;
    const int i   = tid & 15;

    const float cd  = g_card[b * NBINS + i];
    const float ssv = g_ss  [b * NBINS + i];
    s_m [b][i] = (cd > 0.0f) ? ssv / (cd * cd) : 0.0f;
    s_ok[b][i] = (cd > 0.0f) ? 1 : 0;
    __syncthreads();

    float part = 0.0f;
    if (s_ok[b][i]) {
        for (int j = i + 1; j < NBINS; ++j) {
            if (s_ok[b][j]) {
                const float d = sqrtf(s_m[b][i] + s_m[b][j]);
                const float u = 3.0f - d;            // SIGMA_DIS = 3.0
                part += log1pf(u * u);
            }
        }
    }
    s_part[b][i] = part;
    __syncthreads();

    if (i == 0) {
        float sum = 0.0f;
        for (int j = 0; j < NBINS; ++j) sum += s_part[b][j];
        const int nk = g_nk[b];
        const float denom = (float)(nk * (nk - 1));
        s_loss[b] = (nk > 1) ? sum / fmaxf(denom, 1.0f) : 0.0f;
    }
    __syncthreads();

    if (tid == 0) {
        float t = 0.0f;
        for (int j = 0; j < NBATCH; ++j) t += s_loss[j];
        out[0] = t;
    }
}

// ---------------------------------------------------------------------------
__global__ void zero_ws_kernel(unsigned int* __restrict__ p, int nwords)
{
    const int i = blockIdx.x * blockDim.x + threadIdx.x;
    if (i < nwords) p[i] = 0u;
}

extern "C" void kernel_launch(void* const* d_in, const int* in_sizes, int n_in,
                              void* d_out, int out_size, void* d_ws, size_t ws_size,
                              hipStream_t stream)
{
    const float* pred   = (const float*)d_in[0];
    const int*   labels = (const int*)d_in[1];

    float* wss = (float*)d_ws;                       // [16][16] segment sum-of-squares
    float* wcd = wss + NBATCH * NBINS;               // [16][16] cardinality
    int*   wnk = (int*)(wcd + NBATCH * NBINS);       // [16]     max label

    const int nwords = NBATCH * NBINS * 2 + NBATCH;  // 528
    zero_ws_kernel<<<dim3(3), dim3(256), 0, stream>>>((unsigned int*)d_ws, nwords);

    disc_loss_main<<<dim3(BLOCKS_X, NBATCH), dim3(256), 0, stream>>>(
        pred, labels, wss, wcd, wnk);

    disc_loss_finalize<<<dim3(1), dim3(256), 0, stream>>>(
        wss, wcd, wnk, (float*)d_out);
}
